// MultiHeadSelfAttentionModule_44169443672799
// MI455X (gfx1250) — matrix operand order repaired and checked by reading, not compiled, not run on hardware
//
#include <hip/hip_runtime.h>
#include <hip/hip_bf16.h>

// ---------------------------------------------------------------------------
// Problem constants (from reference)
// ---------------------------------------------------------------------------
#define Bn   8
#define Sn   1024
#define Dn   512
#define Hn   8
#define DKn  64
#define SCn  512
#define SKVn 1536           // SC + S

typedef __attribute__((ext_vector_type(16))) _Float16 v16h;
typedef __attribute__((ext_vector_type(2)))  _Float16 v2h;
typedef __attribute__((ext_vector_type(8)))  float    v8f;

// D = A(16x32 f16) * B(32x16 f16) + C(16x16 f32)
__device__ __forceinline__ v8f wmma_f16f32(v16h a, v16h b, v8f c) {
  return __builtin_amdgcn_wmma_f32_16x16x32_f16(false, a, false, b, (short)0, c,
                                                false, false);
}

// ---------------------------------------------------------------------------
// Fragment loaders (layouts per CDNA5 ISA 7.12.2, wave32)
// A 16x32 f16: lane L -> row m = L%16; VGPR i, halves j:
//   i<4 : k = (L/16)*8 + 2i + j          i>=4 : k = 16 + (L/16)*8 + 2(i-4) + j
// B 32x16 f16: lane L -> col n = L%16; VGPR i, halves j: same k as A
// C/D 16x16 f32: VGPR r, lane L -> row = r + 8*(L/16), col = L%16
// ---------------------------------------------------------------------------
__device__ __forceinline__ v16h load_a_f32(const float* __restrict__ X, int ld,
                                           int row, int kb, int lane) {
  int hs = lane >> 4;
  v16h a;
  const float* base = X + row * ld;
#pragma unroll
  for (int i = 0; i < 8; ++i) {
    int k = kb + ((i < 4) ? (hs * 8 + 2 * i) : (16 + hs * 8 + 2 * (i - 4)));
    float2 f = *(const float2*)(base + k);
    a[2 * i]     = (_Float16)f.x;
    a[2 * i + 1] = (_Float16)f.y;
  }
  return a;
}

__device__ __forceinline__ v16h load_a_h(const _Float16* __restrict__ X, int ld,
                                         int row, int kb, int lane) {
  int hs = lane >> 4;
  v16h a;
  const _Float16* base = X + row * ld;
#pragma unroll
  for (int i = 0; i < 8; ++i) {
    int k = kb + ((i < 4) ? (hs * 8 + 2 * i) : (16 + hs * 8 + 2 * (i - 4)));
    v2h p = *(const v2h*)(base + k);
    a[2 * i]     = p[0];
    a[2 * i + 1] = p[1];
  }
  return a;
}

// B operand: logical B[k][n] = M[colrow][kb + kpattern]  (M row-major, ld)
__device__ __forceinline__ v16h load_b_h(const _Float16* __restrict__ M, int ld,
                                         int colrow, int kb, int lane) {
  int hs = lane >> 4;
  v16h b;
  const _Float16* base = M + colrow * ld;
#pragma unroll
  for (int i = 0; i < 8; ++i) {
    int k = kb + ((i < 4) ? (hs * 8 + 2 * i) : (16 + hs * 8 + 2 * (i - 4)));
    v2h p = *(const v2h*)(base + k);
    b[2 * i]     = p[0];
    b[2 * i + 1] = p[1];
  }
  return b;
}

// A fragment from an LDS [16][32] f16 tile (P matrix re-layout)
__device__ __forceinline__ v16h load_a_lds(const _Float16* p, int lane) {
  int hs = lane >> 4, m = lane & 15;
  v16h a;
#pragma unroll
  for (int i = 0; i < 8; ++i) {
    int k = (i < 4) ? (hs * 8 + 2 * i) : (16 + hs * 8 + 2 * (i - 4));
    v2h q = *(const v2h*)(p + m * 32 + k);
    a[2 * i]     = q[0];
    a[2 * i + 1] = q[1];
  }
  return a;
}

// ---------------------------------------------------------------------------
// Kernel 1: fp32 weights -> f16 (Wq|Wk|Wv|Wo concatenated in ws)
// ---------------------------------------------------------------------------
__global__ __launch_bounds__(256) void mhsa_cvt_weights(
    const float* __restrict__ Wq, const float* __restrict__ Wk,
    const float* __restrict__ Wv, const float* __restrict__ Wo,
    _Float16* __restrict__ WH) {
  int tid = blockIdx.x * 256 + threadIdx.x;          // 4 * 262144 total
  int w   = tid >> 18;
  int idx = tid & 262143;
  const float* src = (w == 0) ? Wq : (w == 1) ? Wk : (w == 2) ? Wv : Wo;
  WH[tid] = (_Float16)src[idx];
}

// ---------------------------------------------------------------------------
// Kernel 2: seed new_cache rows [0,SC) from old cache; also emit f16 K rows
// and f16 transposed-V columns for the attention kernel.
// cache layout [B,H,SC,2*DK] fp32 (flat idx == tid).
// ---------------------------------------------------------------------------
__global__ __launch_bounds__(256) void mhsa_seed_cache(
    const float* __restrict__ cache, float* __restrict__ cacheOut,
    _Float16* __restrict__ kh, _Float16* __restrict__ vT) {
  int tid = blockIdx.x * 256 + threadIdx.x;          // B*H*SC*128 total
  int c  = tid & 127;
  int sc = (tid >> 7) & (SCn - 1);
  int h  = (tid >> 16) & (Hn - 1);
  int b  = tid >> 19;
  float v = cache[tid];
  cacheOut[((b * Hn + h) * SKVn + sc) * (2 * DKn) + c] = v;
  if (c < DKn) {
    kh[((b * Hn + h) * SKVn + sc) * DKn + c] = (_Float16)v;
  } else {
    vT[((b * Hn + h) * DKn + (c - DKn)) * SKVn + sc] = (_Float16)v;
  }
}

// ---------------------------------------------------------------------------
// Kernel 3: projection / output GEMM, templated on MODE so the K-loop is
// branch-free.  Y[m,n] = sum_k A[m,k]*W[n,k] + bias[n].
// M = B*S = 8192, N = K = 512.
// One wave computes a 16x64 strip (4 WMMA accumulators): the A fragment is
// loaded once per K-step and reused by 4 B tiles -> 4x arithmetic intensity.
// MODE 0: Y -> qh[B,H,S,DK] (f16)
// MODE 1: Y -> kh rows [SC,SKV) (f16)  +  new_cache K half (fp32)
// MODE 2: Y -> vT cols [SC,SKV) (f16)  +  new_cache V half (fp32)
// MODE 3: A = ctx (f16), Y -> out (fp32)
// ---------------------------------------------------------------------------
template <int MODE>
__global__ __launch_bounds__(256) void mhsa_proj_gemm(
    const float* __restrict__ Xf, const _Float16* __restrict__ Xh,
    const _Float16* __restrict__ WH, const float* __restrict__ bias,
    _Float16* __restrict__ qout, _Float16* __restrict__ kout,
    _Float16* __restrict__ vTout, float* __restrict__ cacheOut,
    float* __restrict__ outF) {
  int lane = threadIdx.x & 31;
  int wave = threadIdx.x >> 5;
  int t  = blockIdx.x * 8 + wave;   // 4096 strips: tm in [0,512), tn in [0,8)
  int tn = t & 7;
  int tm = t >> 3;
  int m0 = tm * 16, n0 = tn * 64;
  int hs = lane >> 4, nn = lane & 15;

  v8f acc[4];
#pragma unroll
  for (int j = 0; j < 4; ++j) {
    float bval = bias[n0 + j * 16 + nn];
#pragma unroll
    for (int r = 0; r < 8; ++r) acc[j][r] = bval;
  }

  for (int kb = 0; kb < Dn; kb += 32) {
    v16h a = (MODE == 3) ? load_a_h(Xh, Dn, m0 + nn, kb, lane)
                         : load_a_f32(Xf, Dn, m0 + nn, kb, lane);
#pragma unroll
    for (int j = 0; j < 4; ++j) {
      v16h bw = load_b_h(WH, Dn, n0 + j * 16 + nn, kb, lane);
      acc[j] = wmma_f16f32(a, bw, acc[j]);
    }
  }

#pragma unroll
  for (int j = 0; j < 4; ++j) {
    int col = n0 + j * 16 + nn;
    int h  = col >> 6;          // col / DK
    int dk = col & (DKn - 1);
#pragma unroll
    for (int r = 0; r < 8; ++r) {
      int row = m0 + r + 8 * hs;
      int b = row >> 10;        // row / S
      int s = row & (Sn - 1);
      float v = acc[j][r];
      if (MODE == 0) {
        qout[((b * Hn + h) * Sn + s) * DKn + dk] = (_Float16)v;
      } else if (MODE == 1) {
        kout[((b * Hn + h) * SKVn + SCn + s) * DKn + dk] = (_Float16)v;
        cacheOut[((b * Hn + h) * SKVn + SCn + s) * (2 * DKn) + dk] = v;
      } else if (MODE == 2) {
        vTout[((b * Hn + h) * DKn + dk) * SKVn + SCn + s] = (_Float16)v;
        cacheOut[((b * Hn + h) * SKVn + SCn + s) * (2 * DKn) + DKn + dk] = v;
      } else {
        outF[row * Dn + col] = v;
      }
    }
  }
}

// ---------------------------------------------------------------------------
// Kernel 4: flash-style attention.  One block = (b, h, 128-query tile);
// each of 8 waves owns 16 query rows, streams 1536 keys in 32-key steps.
// ---------------------------------------------------------------------------
__global__ __launch_bounds__(256) void mhsa_attention(
    const _Float16* __restrict__ qh, const _Float16* __restrict__ kh,
    const _Float16* __restrict__ vT, const int* __restrict__ mask,
    _Float16* __restrict__ ctx) {
  __shared__ _Float16 Plds[8][16 * 32];

  int lane = threadIdx.x & 31;
  int wave = threadIdx.x >> 5;
  int bid = blockIdx.x;            // 512 blocks
  int qt = bid & 7;
  int h  = (bid >> 3) & (Hn - 1);
  int b  = bid >> 6;
  int q0 = qt * 128 + wave * 16;
  int hs = lane >> 4, nn = lane & 15;

  const _Float16* qbh = qh + (b * Hn + h) * Sn * DKn;
  const _Float16* kbh = kh + (b * Hn + h) * SKVn * DKn;
  const _Float16* vbh = vT + (b * Hn + h) * DKn * SKVn;
  const int* mbh = mask + b * Sn * SKVn;

  // Q fragments (dk 0..31 and 32..63) held for the whole key sweep
  v16h aq0 = load_a_h(qbh, DKn, q0 + nn, 0, lane);
  v16h aq1 = load_a_h(qbh, DKn, q0 + nn, 32, lane);

  v8f o0, o1, o2, o3;
  float mrow[8], lrow[8];
#pragma unroll
  for (int r = 0; r < 8; ++r) {
    o0[r] = 0.f; o1[r] = 0.f; o2[r] = 0.f; o3[r] = 0.f;
    mrow[r] = -3.0e38f; lrow[r] = 0.f;
  }

  for (int key0 = 0; key0 < SKVn; key0 += 32) {
    v8f c0, c1;
#pragma unroll
    for (int r = 0; r < 8; ++r) { c0[r] = 0.f; c1[r] = 0.f; }

    // scores = Q . K^T : B[k=dk][n=key] = kh[key0+n][dk]
    v16h bk;
    bk = load_b_h(kbh, DKn, key0 + nn, 0, lane);       c0 = wmma_f16f32(aq0, bk, c0);
    bk = load_b_h(kbh, DKn, key0 + nn, 32, lane);      c0 = wmma_f16f32(aq1, bk, c0);
    bk = load_b_h(kbh, DKn, key0 + 16 + nn, 0, lane);  c1 = wmma_f16f32(aq0, bk, c1);
    bk = load_b_h(kbh, DKn, key0 + 16 + nn, 32, lane); c1 = wmma_f16f32(aq1, bk, c1);

#pragma unroll
    for (int r = 0; r < 8; ++r) {
      int qrow = q0 + r + 8 * hs;
      float s0 = c0[r] * 0.125f;   // 1/sqrt(DK)
      float s1 = c1[r] * 0.125f;
      if (mbh[qrow * SKVn + key0 + nn] == 0)      s0 = -3.0e38f;
      if (mbh[qrow * SKVn + key0 + 16 + nn] == 0) s1 = -3.0e38f;

      float mx = fmaxf(s0, s1);
#pragma unroll
      for (int d = 1; d < 16; d <<= 1) mx = fmaxf(mx, __shfl_xor(mx, d, 32));
      float mnew = fmaxf(mrow[r], mx);
      float corr = __expf(mrow[r] - mnew);
      float p0 = __expf(s0 - mnew);
      float p1 = __expf(s1 - mnew);
      float rs = p0 + p1;
#pragma unroll
      for (int d = 1; d < 16; d <<= 1) rs += __shfl_xor(rs, d, 32);
      lrow[r] = lrow[r] * corr + rs;
      mrow[r] = mnew;
      o0[r] *= corr; o1[r] *= corr; o2[r] *= corr; o3[r] *= corr;

      int prow = r + 8 * hs;       // C-layout row -> LDS [16][32] f16
      Plds[wave][prow * 32 + nn]      = (_Float16)p0;
      Plds[wave][prow * 32 + 16 + nn] = (_Float16)p1;
    }

    // P(16x32) . V(32x64): B[k=key][n=dk] = vT[dkBase+n][key0+k]
    v16h pa = load_a_lds(&Plds[wave][0], lane);
    v16h bv;
    bv = load_b_h(vbh, SKVn,  0 + nn, key0, lane); o0 = wmma_f16f32(pa, bv, o0);
    bv = load_b_h(vbh, SKVn, 16 + nn, key0, lane); o1 = wmma_f16f32(pa, bv, o1);
    bv = load_b_h(vbh, SKVn, 32 + nn, key0, lane); o2 = wmma_f16f32(pa, bv, o2);
    bv = load_b_h(vbh, SKVn, 48 + nn, key0, lane); o3 = wmma_f16f32(pa, bv, o3);
  }

#pragma unroll
  for (int r = 0; r < 8; ++r) {
    float inv = 1.0f / lrow[r];
    int s = q0 + r + 8 * hs;
    _Float16* dst = ctx + (b * Sn + s) * Dn + h * DKn;
    dst[ 0 + nn] = (_Float16)(o0[r] * inv);
    dst[16 + nn] = (_Float16)(o1[r] * inv);
    dst[32 + nn] = (_Float16)(o2[r] * inv);
    dst[48 + nn] = (_Float16)(o3[r] * inv);
  }
}

// ---------------------------------------------------------------------------
// Host-side orchestration
// ---------------------------------------------------------------------------
extern "C" void kernel_launch(void* const* d_in, const int* in_sizes, int n_in,
                              void* d_out, int out_size, void* d_ws,
                              size_t ws_size, hipStream_t stream) {
  (void)in_sizes; (void)n_in; (void)out_size; (void)ws_size;
  const float* query = (const float*)d_in[0];
  const float* key   = (const float*)d_in[1];
  const float* value = (const float*)d_in[2];
  const int*   amask = (const int*)  d_in[3];
  const float* cache = (const float*)d_in[4];
  const float* Wq = (const float*)d_in[5];
  const float* bq = (const float*)d_in[6];
  const float* Wk = (const float*)d_in[7];
  const float* bk = (const float*)d_in[8];
  const float* Wv = (const float*)d_in[9];
  const float* bv = (const float*)d_in[10];
  const float* Wo = (const float*)d_in[11];
  const float* bo = (const float*)d_in[12];

  float* outF     = (float*)d_out;                        // [B,S,D]
  float* cacheOut = outF + (size_t)Bn * Sn * Dn;          // [B,H,SKV,2*DK]

  // workspace (f16 staging; whole pipeline fits in 192MB L2)
  _Float16* ws = (_Float16*)d_ws;
  _Float16* WH  = ws;                                     // 4 * 512*512
  _Float16* QH  = WH  + (size_t)4 * Dn * Dn;              // [B,H,S,DK]
  _Float16* KH  = QH  + (size_t)Bn * Hn * Sn * DKn;       // [B,H,SKV,DK]
  _Float16* VT  = KH  + (size_t)Bn * Hn * SKVn * DKn;     // [B,H,DK,SKV]
  _Float16* CTX = VT  + (size_t)Bn * Hn * DKn * SKVn;     // [B,S,D]

  // 1) weights -> f16
  mhsa_cvt_weights<<<(4 * Dn * Dn) / 256, 256, 0, stream>>>(Wq, Wk, Wv, Wo, WH);

  // 2) old cache -> new_cache rows [0,SC) + f16 K rows / transposed-V cols
  mhsa_seed_cache<<<(Bn * Hn * SCn * 2 * DKn) / 256, 256, 0, stream>>>(
      cache, cacheOut, KH, VT);

  const int gemm_blocks = (Bn * Sn / 16) * (Dn / 64) / 8;  // 512

  // 3) Q/K/V projections
  mhsa_proj_gemm<0><<<gemm_blocks, 256, 0, stream>>>(
      query, nullptr, WH + 0 * Dn * Dn, bq, QH, KH, VT, cacheOut, outF);
  mhsa_proj_gemm<1><<<gemm_blocks, 256, 0, stream>>>(
      key, nullptr, WH + 1 * Dn * Dn, bk, QH, KH, VT, cacheOut, outF);
  mhsa_proj_gemm<2><<<gemm_blocks, 256, 0, stream>>>(
      value, nullptr, WH + 2 * Dn * Dn, bv, QH, KH, VT, cacheOut, outF);

  // 4) attention
  mhsa_attention<<<Bn * Hn * (Sn / 128), 256, 0, stream>>>(QH, KH, VT, amask, CTX);

  // 5) output projection
  mhsa_proj_gemm<3><<<gemm_blocks, 256, 0, stream>>>(
      nullptr, CTX, WH + 3 * Dn * Dn, bo, QH, KH, VT, cacheOut, outF);
}
